// LinearMemoryAttention_970662609004
// MI455X (gfx1250) — compile-verified, matrix operand order checked by reading, not run on hardware
//
#include <hip/hip_runtime.h>
#include <stdint.h>

typedef __bf16 bf16;
typedef __attribute__((ext_vector_type(16))) __bf16 v16bf;
typedef __attribute__((ext_vector_type(8)))  __bf16 v8bf;
typedef __attribute__((ext_vector_type(8)))  float  v8f;

#define LDB 72            // padded bf16 row stride (144B, 16B aligned, breaks bank cycles)
#define EPS_F 1e-6f
#define S_LEN 4096
#define H_NUM 16

__device__ __forceinline__ float phi1(float x) {
  // elu(x)+1 == x+1 (x>0) else exp(x)
  return x > 0.f ? x + 1.f : __expf(x);
}

// Load a 16x32 bf16 WMMA operand fragment (A or B) from an LDS row whose
// 16 "rows" are indexed by (lane&15) externally. Per ISA 7.12.2:
// elements 0..7 -> K = k0 + 8*(lane>=16) + j ; elements 8..15 -> +16 more.
__device__ __forceinline__ v16bf ldfrag(const bf16* rowbase, int k0hi) {
  const bf16* p = rowbase + k0hi;
  v8bf lo = *(const v8bf*)(p);
  v8bf hi = *(const v8bf*)(p + 16);
  return __builtin_shufflevector(lo, hi, 0,1,2,3,4,5,6,7,8,9,10,11,12,13,14,15);
}

__device__ __forceinline__ v8f wmma_bf16(v16bf a, v16bf b, v8f c) {
  return __builtin_amdgcn_wmma_f32_16x16x32_bf16(false, a, false, b, (short)0, c, false, false);
}

__device__ __forceinline__ v8f zero8() {
  v8f z;
#pragma unroll
  for (int i = 0; i < 8; ++i) z[i] = 0.f;
  return z;
}

// Issue 4 async B128 loads (16 floats) from global (GVS: uniform SGPR base +
// per-lane byte offset) into LDS. Tracked by ASYNCcnt.
__device__ __forceinline__ void async_slice(const float* __restrict__ sbase,
                                            uint32_t lds_base, int elem_off) {
#pragma unroll
  for (int jj = 0; jj < 16; jj += 4) {
    const uint32_t l = lds_base + (uint32_t)((elem_off + jj) * 4);
    const uint32_t g = (uint32_t)((elem_off + jj) * 4);
    asm volatile("global_load_async_to_lds_b128 %0, %1, %2 offset:0"
                 :: "v"(l), "v"(g), "s"(sbase) : "memory");
  }
}

__global__ __launch_bounds__(256)
void LinearMemoryAttention_chunked_kernel(const float* __restrict__ qp,
                                          const float* __restrict__ kp,
                                          const float* __restrict__ vp,
                                          const int*   __restrict__ maskp,
                                          const float* __restrict__ gatep,
                                          float*       __restrict__ outp) {
  // bf16 WMMA operand staging
  __shared__ __align__(16) bf16 PhiQ [64 * LDB];   // [pos][d]
  __shared__ __align__(16) bf16 PhiK [64 * LDB];   // [pos][d]
  __shared__ __align__(16) bf16 PhiKT[64 * LDB];   // [d][pos]
  __shared__ __align__(16) bf16 VgT  [64 * LDB];   // [e][pos]
  __shared__ __align__(16) bf16 Slds [64 * LDB];   // [m][n] masked scores (bf16)
  __shared__ __align__(16) bf16 KVbT [64 * LDB];   // [e][d] bf16 copy of KV state
  // raw fp32 double-buffered async landing zone (96 KB)
  __shared__ __align__(16) float Qraw[2][64 * 64];
  __shared__ __align__(16) float Kraw[2][64 * 64];
  __shared__ __align__(16) float Vraw[2][64 * 64];
  __shared__ float kstate[64];
  __shared__ float rdenS[64];

  const int tid  = threadIdx.x;
  const int bh   = blockIdx.x;        // 0..63 (b*H + h)
  const int b    = bh >> 4;           // H = 16
  const int lane = tid & 31;
  const int wave = tid >> 5;          // 0..7
  const int ms   = wave & 3;          // M-strip (rows m0..m0+15)
  const int eh   = wave >> 2;         // e half (0: cols 0..31, 1: 32..63)
  const int m0   = ms << 4;
  const int l15  = lane & 15;
  const int hi8  = (lane >> 4) << 3;  // 8*(lane>=16)

  // per-thread staging slice: 16 floats of row `row` per array
  const int row = tid >> 2;           // 0..63
  const int c0  = (tid & 3) << 4;     // 0,16,32,48
  const int eo  = row * 64 + c0;      // element offset within a chunk

  const uint32_t lQ[2] = {(uint32_t)(uintptr_t)&Qraw[0][0], (uint32_t)(uintptr_t)&Qraw[1][0]};
  const uint32_t lK[2] = {(uint32_t)(uintptr_t)&Kraw[0][0], (uint32_t)(uintptr_t)&Kraw[1][0]};
  const uint32_t lV[2] = {(uint32_t)(uintptr_t)&Vraw[0][0], (uint32_t)(uintptr_t)&Vraw[1][0]};

  // ---- init recurrent state ----
  for (int i = tid; i < 64 * LDB; i += 256) KVbT[i] = (bf16)0.f;
  if (tid < 64) kstate[tid] = 0.f;

  v8f kvacc[4];                       // KV fp32 accumulators (used by eh==1 waves)
#pragma unroll
  for (int t = 0; t < 4; ++t) kvacc[t] = zero8();

  // kick off async staging of chunk 0 into buffer 0
  {
    const long nb0 = (long)bh * S_LEN * 64;
    async_slice(qp + nb0, lQ[0], eo);
    async_slice(kp + nb0, lK[0], eo);
    async_slice(vp + nb0, lV[0], eo);
  }

  __syncthreads();

  for (int ch = 0; ch < 64; ++ch) {
    const int buf = ch & 1;
    // ============ stage A: prefetch next chunk, featurize current ===========
    if (ch < 63) {                    // issue chunk ch+1 into the other buffer
      const long nb = ((long)bh * S_LEN + ((ch + 1) << 6)) * 64;
      const int nbuf = buf ^ 1;
      async_slice(qp + nb, lQ[nbuf], eo);
      async_slice(kp + nb, lK[nbuf], eo);
      async_slice(vp + nb, lV[nbuf], eo);
      asm volatile("s_wait_asynccnt 0xc" ::: "memory");  // retire chunk-ch batch
    } else {
      asm volatile("s_wait_asynccnt 0x0" ::: "memory");
    }
    {
      const int pos = (ch << 6) + row;
      const long goff = (long)bh * S_LEN + pos;
      const float gv = gatep[goff];
      const float tm = (float)maskp[(long)b * S_LEN + pos];
      const float gm = gv * tm;
      const float* __restrict__ Qr = &Qraw[buf][0];
      const float* __restrict__ Kr = &Kraw[buf][0];
      const float* __restrict__ Vr = &Vraw[buf][0];
#pragma unroll
      for (int jj = 0; jj < 16; jj += 4) {
        const float4 qv = *(const float4*)(Qr + eo + jj);
        const float4 kv = *(const float4*)(Kr + eo + jj);
        const float4 vv = *(const float4*)(Vr + eo + jj);
        const float qa[4] = {qv.x, qv.y, qv.z, qv.w};
        const float ka[4] = {kv.x, kv.y, kv.z, kv.w};
        const float va[4] = {vv.x, vv.y, vv.z, vv.w};
#pragma unroll
        for (int u = 0; u < 4; ++u) {
          const int c = c0 + jj + u;
          const float fq = phi1(qa[u]) * tm;
          const float fk = phi1(ka[u]) * gm;
          const float fv = va[u] * gm;
          PhiQ [row * LDB + c]   = (bf16)fq;
          PhiK [row * LDB + c]   = (bf16)fk;
          PhiKT[c   * LDB + row] = (bf16)fk;
          VgT  [c   * LDB + row] = (bf16)fv;
        }
      }
    }
    __syncthreads();   // (A) chunk operands + previous-chunk KVbT/kstate ready

    // ================= phase 1 ==============================================
    // out_inter = PhiQ @ KV_prev  (all waves, 2 e-tiles each)
    v8f acc[2] = {zero8(), zero8()};
    const bf16* arow = &PhiQ[(m0 + l15) * LDB];
#pragma unroll
    for (int et = 0; et < 2; ++et) {
      const bf16* brow = &KVbT[(eh * 32 + et * 16 + l15) * LDB];
#pragma unroll
      for (int kt = 0; kt < 2; ++kt)
        acc[et] = wmma_bf16(ldfrag(arow, kt * 32 + hi8),
                            ldfrag(brow, kt * 32 + hi8), acc[et]);
    }

    // den_inter = phi_q . k_state_prev (consumed before kstate is updated)
    float dend = 0.f;
    if (tid < 64) {
      for (int d = 0; d < 64; ++d)
        dend += (float)PhiQ[tid * LDB + d] * kstate[d];
    }

    if (eh == 1) {
      // KV += PhiK^T @ Vg   (wave owns d-strip ms, all 4 e-tiles)
      const bf16* krow = &PhiKT[(m0 + l15) * LDB];
#pragma unroll
      for (int et = 0; et < 4; ++et) {
        const bf16* brow = &VgT[(et * 16 + l15) * LDB];
#pragma unroll
        for (int kt = 0; kt < 2; ++kt)
          kvacc[et] = wmma_bf16(ldfrag(krow, kt * 32 + hi8),
                                ldfrag(brow, kt * 32 + hi8), kvacc[et]);
      }
    } else {
      // S = tril_incl(PhiQ @ PhiK^T), stored bf16 to LDS (zeros above diag)
#pragma unroll
      for (int nt = 0; nt < 4; ++nt) {
        if (nt <= ms) {
          v8f s = zero8();
          const bf16* brow = &PhiK[(nt * 16 + l15) * LDB];
#pragma unroll
          for (int kt = 0; kt < 2; ++kt)
            s = wmma_bf16(ldfrag(arow, kt * 32 + hi8),
                          ldfrag(brow, kt * 32 + hi8), s);
#pragma unroll
          for (int r = 0; r < 8; ++r) {
            float sv = s[r];
            if (nt == ms && l15 > r + hi8) sv = 0.f;   // inclusive causal mask
            Slds[(m0 + r + hi8) * LDB + nt * 16 + l15] = (bf16)sv;
          }
        } else {
#pragma unroll
          for (int r = 0; r < 8; ++r)
            Slds[(m0 + r + hi8) * LDB + nt * 16 + l15] = (bf16)0.f;
        }
      }
    }
    __syncthreads();   // (B) S ready; out_inter done -> KVbT free to overwrite

    // ================= phase 2: denom, state updates ========================
    if (tid < 64) {
      float rs = 0.f;
      for (int n = 0; n < 64; ++n) rs += (float)Slds[tid * LDB + n];
      rdenS[tid] = 1.f / fmaxf(dend + rs, EPS_F);
    } else if (tid < 128) {
      const int d = tid - 64;
      float cs = 0.f;
      for (int n = 0; n < 64; ++n) cs += (float)PhiKT[d * LDB + n];
      kstate[d] += cs;
    }
    if (eh == 1) {   // publish bf16 KV state for next chunk's out_inter
#pragma unroll
      for (int et = 0; et < 4; ++et)
#pragma unroll
        for (int r = 0; r < 8; ++r)
          KVbT[(et * 16 + l15) * LDB + m0 + hi8 + r] = (bf16)kvacc[et][r];
    }
    __syncthreads();   // (C) rden / KVbT / kstate published

    // ================= phase 3: out_intra = S @ Vg, scale, store ============
    const bf16* srow = &Slds[(m0 + l15) * LDB];
#pragma unroll
    for (int et = 0; et < 2; ++et) {
      const bf16* brow = &VgT[(eh * 32 + et * 16 + l15) * LDB];
#pragma unroll
      for (int kt = 0; kt < 2; ++kt)
        acc[et] = wmma_bf16(ldfrag(srow, kt * 32 + hi8),
                            ldfrag(brow, kt * 32 + hi8), acc[et]);
    }
    float rd[8];
#pragma unroll
    for (int r = 0; r < 8; ++r) rd[r] = rdenS[m0 + r + hi8];
#pragma unroll
    for (int et = 0; et < 2; ++et) {
#pragma unroll
      for (int r = 0; r < 8; ++r) {
        const long o = ((long)bh * S_LEN + (ch << 6) + m0 + r + hi8) * 64
                     + eh * 32 + et * 16 + l15;
        outp[o] = acc[et][r] * rd[r];
      }
    }
    __syncthreads();   // (D) all LDS consumers done before next stage A
  }
}

__global__ __launch_bounds__(256)
void GateMean_kernel(const float* __restrict__ gp, float* __restrict__ o, int N) {
  __shared__ float red[256];
  float s = 0.f;
  for (int i = threadIdx.x; i < N; i += 256) s += gp[i];
  red[threadIdx.x] = s;
  __syncthreads();
  for (int off = 128; off > 0; off >>= 1) {
    if ((int)threadIdx.x < off) red[threadIdx.x] += red[threadIdx.x + off];
    __syncthreads();
  }
  if (threadIdx.x == 0) o[0] = red[0] / (float)N;
}

extern "C" void kernel_launch(void* const* d_in, const int* in_sizes, int n_in,
                              void* d_out, int out_size, void* d_ws, size_t ws_size,
                              hipStream_t stream) {
  const float* q    = (const float*)d_in[0];
  const float* k    = (const float*)d_in[1];
  const float* v    = (const float*)d_in[2];
  const int*   mask = (const int*)  d_in[3];
  const float* gate = (const float*)d_in[4];
  float* out = (float*)d_out;

  LinearMemoryAttention_chunked_kernel<<<dim3(64), dim3(256), 0, stream>>>(
      q, k, v, mask, gate, out);

  const long NG = 4L * H_NUM * S_LEN;             // B*H*S gate elements
  GateMean_kernel<<<dim3(1), dim3(256), 0, stream>>>(
      gate, out + 4L * H_NUM * S_LEN * 64, (int)NG);

  (void)in_sizes; (void)n_in; (void)out_size; (void)d_ws; (void)ws_size;
}
